// FloodGNN_29386166239380
// MI455X (gfx1250) — compile-verified, matrix operand order checked by reading, not run on hardware
//
#include <hip/hip_runtime.h>
#include <hip/hip_bf16.h>

typedef __attribute__((ext_vector_type(16))) _Float16 v16h;
typedef __attribute__((ext_vector_type(8)))  _Float16 v8h;
typedef __attribute__((ext_vector_type(2)))  _Float16 v2h;
typedef __attribute__((ext_vector_type(8)))  float    v8f;

#define HDIM   64
#define HHALF  32          // HDIM/2
#define EATTR  3
#define KIN    67          // HDIM + EATTR
#define KPAD   96          // 3 chunks of 32
#define NLAYER 3

// K index of half-pair (v,p) for lanes 0-15 of the CDNA5 16-bit 16x32 A/B layout.
// Lanes 16-31 add +8 (hi*8).  VGPR0..3 -> K 0..7 ; VGPR4..7 -> K 16..23.
__device__ __forceinline__ int wmma_kpair(int v, int p) {
    return ((v < 4) ? (2 * v) : (8 + 2 * v)) + p;
}

static __device__ __forceinline__ v8f wmma_f16(v16h a, v16h b, v8f c) {
    return __builtin_amdgcn_wmma_f32_16x16x32_f16(false, a, false, b, (short)0, c,
                                                  false, false);
}

// ---------------------------------------------------------------------------
// Edge-MLP kernel: per wave, tiles of 16 edges.
//   m = relu([h[src]|attr] @ W1 + b1) @ W2 + b2 ; agg[dst] += m * inv_deg[dst]
// GEMM1: 16xKPAD @ KPADx32  (3 K-chunks x 2 N-tiles = 6 WMMAs)
// GEMM2: 16x32   @ 32x64    (4 N-tiles            = 4 WMMAs)
// Biases are pre-seeded into the WMMA C operand (column-constant broadcast).
// ---------------------------------------------------------------------------
__global__ __launch_bounds__(128) void edge_mlp_kernel(
    const float* __restrict__ hbuf,   // [N,64]
    const int*   __restrict__ ei,     // [2,E]  (src row, dst row)
    const float* __restrict__ eattr,  // [E,3]
    const float* __restrict__ w1,     // [67,32]
    const float* __restrict__ b1,     // [32]
    const float* __restrict__ w2,     // [32,64]
    const float* __restrict__ b2,     // [64]
    const float* __restrict__ invc,   // [N]
    float*       __restrict__ agg,    // [N,64]
    int E_, int n_tiles)
{
    __shared__ alignas(16) _Float16 ldsA[4][16][KPAD];   // staged A (f16)
    __shared__ alignas(16) _Float16 ldsH[4][16][HHALF];  // hidden (f16)
    __shared__ int   ldsD[4][16];
    __shared__ float ldsS[4][16];

    const int tid  = threadIdx.x;
    const int wv   = tid >> 5;
    const int lane = tid & 31;
    const int l    = lane & 15;
    const int hi   = lane >> 4;

    // ---- per-layer constant B fragments, held in registers ----
    v16h Bw1[3][2];
#pragma unroll
    for (int kc = 0; kc < 3; ++kc)
#pragma unroll
        for (int nt = 0; nt < 2; ++nt)
#pragma unroll
            for (int v = 0; v < 8; ++v)
#pragma unroll
                for (int p = 0; p < 2; ++p) {
                    int k = kc * 32 + wmma_kpair(v, p) + hi * 8;
                    Bw1[kc][nt][2 * v + p] =
                        (k < KIN) ? (_Float16)w1[k * HHALF + nt * 16 + l]
                                  : (_Float16)0.0f;
                }
    v16h Bw2[4];
#pragma unroll
    for (int nt = 0; nt < 4; ++nt)
#pragma unroll
        for (int v = 0; v < 8; ++v)
#pragma unroll
            for (int p = 0; p < 2; ++p) {
                int k = wmma_kpair(v, p) + hi * 8;
                Bw2[nt][2 * v + p] = (_Float16)w2[k * HDIM + nt * 16 + l];
            }

    // broadcast-bias accumulator seeds (column-constant over the 8 rows/lane)
    v8f bias1[2], bias2[4];
#pragma unroll
    for (int i = 0; i < 8; ++i) {
        bias1[0][i] = b1[l];
        bias1[1][i] = b1[16 + l];
        bias2[0][i] = b2[l];
        bias2[1][i] = b2[16 + l];
        bias2[2][i] = b2[32 + l];
        bias2[3][i] = b2[48 + l];
    }

    // one-time zero of constant pad columns 67..95 (never rewritten per tile)
#pragma unroll
    for (int r = 0; r < 16; ++r)
        if (lane >= EATTR) ldsA[wv][r][64 + lane] = (_Float16)0.0f;
    __builtin_amdgcn_wave_barrier();

    const int stride = gridDim.x * 4;
    for (int tile = blockIdx.x * 4 + wv; tile < n_tiles; tile += stride) {
        const int ebase = tile * 16;
        if (tile + stride < n_tiles)
            __builtin_prefetch(&ei[(tile + stride) * 16], 0, 1);

        // ---- gather dst / inv-degree for this tile ----
        if (lane < 16) {
            int e = ebase + lane;
            int d = (e < E_) ? ei[E_ + e] : -1;
            ldsD[wv][lane] = d;
            ldsS[wv][lane] = (d >= 0) ? invc[d] : 0.0f;
        }
        // ---- gather A tile: rows = edges, cols = [h[src] (64) | attr (3)] ----
        // lane covers h columns 2*lane, 2*lane+1  -> one b64 load, one b32 store
#pragma unroll 4
        for (int r = 0; r < 16; ++r) {
            int e = ebase + r;
            float2 hv = make_float2(0.0f, 0.0f);
            float  av = 0.0f;
            if (e < E_) {
                int s = ei[e];
                hv = *(const float2*)(hbuf + (size_t)s * HDIM + 2 * lane);
                if (lane < EATTR) av = eattr[e * EATTR + lane];
            }
            v2h pk = { (_Float16)hv.x, (_Float16)hv.y };
            *(v2h*)&ldsA[wv][r][2 * lane] = pk;
            if (lane < EATTR) ldsA[wv][r][64 + lane] = (_Float16)av;
        }
        __builtin_amdgcn_wave_barrier();   // same-wave LDS RAW; keep sched order

        // ---- GEMM1 (bias seeded into C of first chunk) ----
        v8f acc0 = bias1[0], acc1 = bias1[1];
#pragma unroll
        for (int kc = 0; kc < 3; ++kc) {
            const _Float16* ap = &ldsA[wv][l][kc * 32 + hi * 8];
            v8h alo = *(const v8h*)ap;          // K = hi*8 + 0..7
            v8h ahi = *(const v8h*)(ap + 16);   // K = 16 + hi*8 + 0..7
            v16h a  = __builtin_shufflevector(alo, ahi,
                        0,1,2,3,4,5,6,7,8,9,10,11,12,13,14,15);
            acc0 = wmma_f16(a, Bw1[kc][0], acc0);
            acc1 = wmma_f16(a, Bw1[kc][1], acc1);
        }
        // ---- relu, restage as A for GEMM2 (C-layout -> A-layout) ----
#pragma unroll
        for (int r = 0; r < 8; ++r) {
            int m = r + hi * 8;
            ldsH[wv][m][l]      = (_Float16)fmaxf(acc0[r], 0.0f);
            ldsH[wv][m][16 + l] = (_Float16)fmaxf(acc1[r], 0.0f);
        }
        __builtin_amdgcn_wave_barrier();

        // ---- GEMM2 (bias seeded into C) ----
        const _Float16* hp = &ldsH[wv][l][hi * 8];
        v8h hlo = *(const v8h*)hp;
        v8h hh2 = *(const v8h*)(hp + 16);
        v16h a2 = __builtin_shufflevector(hlo, hh2,
                    0,1,2,3,4,5,6,7,8,9,10,11,12,13,14,15);
        v8f c0 = wmma_f16(a2, Bw2[0], bias2[0]);
        v8f c1 = wmma_f16(a2, Bw2[1], bias2[1]);
        v8f c2 = wmma_f16(a2, Bw2[2], bias2[2]);
        v8f c3 = wmma_f16(a2, Bw2[3], bias2[3]);

        // ---- mean-aggregate scatter: agg[dst] += m * inv_deg[dst] ----
#pragma unroll
        for (int r = 0; r < 8; ++r) {
            int m = r + hi * 8;
            int d = ldsD[wv][m];
            if (d >= 0) {
                float sc = ldsS[wv][m];
                float* base = agg + (size_t)d * HDIM;
                atomicAdd(base + l,      c0[r] * sc);
                atomicAdd(base + 16 + l, c1[r] * sc);
                atomicAdd(base + 32 + l, c2[r] * sc);
                atomicAdd(base + 48 + l, c3[r] * sc);
            }
        }
    }
}

// ---------------------------------------------------------------------------
// Encoder: h = relu(LN(x @ W + b)); 64 threads per node, 4 nodes per block.
// ---------------------------------------------------------------------------
__global__ __launch_bounds__(256) void encoder_kernel(
    const float* __restrict__ x, const float* __restrict__ W,
    const float* __restrict__ bias, const float* __restrict__ g,
    const float* __restrict__ beta, float* __restrict__ hout, int N_)
{
    __shared__ float s1[256], s2[256];
    const int tid  = threadIdx.x;
    const int node = blockIdx.x * 4 + (tid >> 6);
    const int c    = tid & 63;
    float v = bias[c];
    if (node < N_) {
#pragma unroll
        for (int k = 0; k < 16; ++k) v += x[node * 16 + k] * W[k * HDIM + c];
    }
    s1[tid] = v; s2[tid] = v * v;
    __syncthreads();
#pragma unroll
    for (int off = 32; off >= 1; off >>= 1) {
        if (c < off) { s1[tid] += s1[tid + off]; s2[tid] += s2[tid + off]; }
        __syncthreads();
    }
    const int base = tid & ~63;
    float mu  = s1[base] * (1.0f / 64.0f);
    float var = s2[base] * (1.0f / 64.0f) - mu * mu;
    float rs  = rsqrtf(var + 1e-5f);
    float o   = fmaxf((v - mu) * rs * g[c] + beta[c], 0.0f);
    if (node < N_) hout[node * HDIM + c] = o;
}

// h = LN(h + agg)  (no relu)
__global__ __launch_bounds__(256) void update_kernel(
    float* __restrict__ hbuf, const float* __restrict__ agg,
    const float* __restrict__ g, const float* __restrict__ beta, int N_)
{
    __shared__ float s1[256], s2[256];
    const int tid  = threadIdx.x;
    const int node = blockIdx.x * 4 + (tid >> 6);
    const int c    = tid & 63;
    float v = 0.0f;
    if (node < N_) v = hbuf[node * HDIM + c] + agg[node * HDIM + c];
    s1[tid] = v; s2[tid] = v * v;
    __syncthreads();
#pragma unroll
    for (int off = 32; off >= 1; off >>= 1) {
        if (c < off) { s1[tid] += s1[tid + off]; s2[tid] += s2[tid + off]; }
        __syncthreads();
    }
    const int base = tid & ~63;
    float mu  = s1[base] * (1.0f / 64.0f);
    float var = s2[base] * (1.0f / 64.0f) - mu * mu;
    float rs  = rsqrtf(var + 1e-5f);
    float o   = (v - mu) * rs * g[c] + beta[c];
    if (node < N_) hbuf[node * HDIM + c] = o;
}

__global__ void count_kernel(const int* __restrict__ ei, float* __restrict__ cnt, int E_) {
    int e = blockIdx.x * 256 + threadIdx.x;
    if (e < E_) atomicAdd(&cnt[ei[E_ + e]], 1.0f);
}

__global__ void invcnt_kernel(float* __restrict__ cnt, int N_) {
    int i = blockIdx.x * 256 + threadIdx.x;
    if (i < N_) { float c = cnt[i]; cnt[i] = (c > 0.0f) ? (1.0f / c) : 0.0f; }
}

// depth = relu(h@dw1+db1)@dw2+db2 ; vel = relu(h@vw1+vb1)@vw2+vb2
__global__ __launch_bounds__(256) void heads_kernel(
    const float* __restrict__ hbuf,
    const float* __restrict__ dw1, const float* __restrict__ db1,
    const float* __restrict__ dw2, const float* __restrict__ db2,
    const float* __restrict__ vw1, const float* __restrict__ vb1,
    const float* __restrict__ vw2, const float* __restrict__ vb2,
    float* __restrict__ out, int N_)
{
    int n = blockIdx.x * 256 + threadIdx.x;
    if (n >= N_) return;
    float hv[HDIM];
#pragma unroll
    for (int k = 0; k < HDIM; ++k) hv[k] = hbuf[(size_t)n * HDIM + k];

    float acc = db2[0];
    for (int j = 0; j < HHALF; ++j) {
        float s = db1[j];
#pragma unroll
        for (int k = 0; k < HDIM; ++k) s += hv[k] * dw1[k * HHALF + j];
        acc += fmaxf(s, 0.0f) * dw2[j];
    }
    out[n] = acc;

    float a0 = vb2[0], a1 = vb2[1];
    for (int j = 0; j < HHALF; ++j) {
        float s = vb1[j];
#pragma unroll
        for (int k = 0; k < HDIM; ++k) s += hv[k] * vw1[k * HHALF + j];
        float rl = fmaxf(s, 0.0f);
        a0 += rl * vw2[j * 2];
        a1 += rl * vw2[j * 2 + 1];
    }
    out[N_ + n * 2]     = a0;
    out[N_ + n * 2 + 1] = a1;
}

extern "C" void kernel_launch(void* const* d_in, const int* in_sizes, int n_in,
                              void* d_out, int out_size, void* d_ws, size_t ws_size,
                              hipStream_t stream)
{
    const float* x      = (const float*)d_in[0];
    const int*   ei     = (const int*)  d_in[1];
    const float* eattr  = (const float*)d_in[2];
    const float* enc_w  = (const float*)d_in[3];
    const float* enc_b  = (const float*)d_in[4];
    const float* enc_g  = (const float*)d_in[5];
    const float* enc_be = (const float*)d_in[6];
    const float* mlp_w1 = (const float*)d_in[7];
    const float* mlp_b1 = (const float*)d_in[8];
    const float* mlp_w2 = (const float*)d_in[9];
    const float* mlp_b2 = (const float*)d_in[10];
    const float* ln_g   = (const float*)d_in[11];
    const float* ln_b   = (const float*)d_in[12];
    const float* dw1    = (const float*)d_in[13];
    const float* db1    = (const float*)d_in[14];
    const float* dw2    = (const float*)d_in[15];
    const float* db2    = (const float*)d_in[16];
    const float* vw1    = (const float*)d_in[17];
    const float* vb1    = (const float*)d_in[18];
    const float* vw2    = (const float*)d_in[19];
    const float* vb2    = (const float*)d_in[20];

    const int N_ = in_sizes[0] / 16;
    const int E_ = in_sizes[1] / 2;

    // workspace: h[N*64] | agg[N*64] | deg/inv_deg[N]   (~25.8 MB for N=50k)
    float* hbuf = (float*)d_ws;
    float* agg  = hbuf + (size_t)N_ * HDIM;
    float* cnt  = agg  + (size_t)N_ * HDIM;

    hipMemsetAsync(cnt, 0, (size_t)N_ * sizeof(float), stream);
    encoder_kernel<<<(N_ + 3) / 4, 256, 0, stream>>>(x, enc_w, enc_b, enc_g, enc_be,
                                                     hbuf, N_);
    count_kernel<<<(E_ + 255) / 256, 256, 0, stream>>>(ei, cnt, E_);
    invcnt_kernel<<<(N_ + 255) / 256, 256, 0, stream>>>(cnt, N_);

    const int n_tiles = (E_ + 15) / 16;
    for (int i = 0; i < NLAYER; ++i) {
        hipMemsetAsync(agg, 0, (size_t)N_ * HDIM * sizeof(float), stream);
        edge_mlp_kernel<<<(n_tiles + 3) / 4, 128, 0, stream>>>(
            hbuf, ei, eattr,
            mlp_w1 + (size_t)i * KIN * HHALF, mlp_b1 + (size_t)i * HHALF,
            mlp_w2 + (size_t)i * HHALF * HDIM, mlp_b2 + (size_t)i * HDIM,
            cnt, agg, E_, n_tiles);
        update_kernel<<<(N_ + 3) / 4, 256, 0, stream>>>(
            hbuf, agg, ln_g + (size_t)i * HDIM, ln_b + (size_t)i * HDIM, N_);
    }
    heads_kernel<<<(N_ + 255) / 256, 256, 0, stream>>>(
        hbuf, dw1, db1, dw2, db2, vw1, vb1, vw2, vb2, (float*)d_out, N_);
}
